// PitchLoss_61761629717079
// MI455X (gfx1250) — compile-verified, matrix operand order checked by reading, not run on hardware
//
#include <hip/hip_runtime.h>

#define T_LEN   32768
#define NN      1024
#define TOTAL_F 33554432.0f   // T * N_NOTES

typedef __attribute__((ext_vector_type(16))) _Float16 v16h;
typedef __attribute__((ext_vector_type(8)))  float    v8f;

// ---------------------------------------------------------------------------
// prep1: one workgroup, 1024 threads, 32 elems/thread.
//   S[k]       = # of nonzero onsets with index < k   (k = 0..T)   [float]
//   off_idx[r] = index of r-th nonzero offset, fill T-1            [int]
// Hillis-Steele block scan in LDS over per-thread chunk totals.
// ---------------------------------------------------------------------------
__global__ __launch_bounds__(1024) void pl_prep1(const float* __restrict__ onsets,
                                                 const float* __restrict__ offsets,
                                                 float* __restrict__ S,
                                                 int* __restrict__ off_idx) {
    __shared__ float sh[1024];
    const int tid  = threadIdx.x;
    const int base = tid * 32;

    // ---- scan onsets indicator -> S ----
    float s0 = 0.f;
    #pragma unroll 8
    for (int k = 0; k < 32; ++k) s0 += (onsets[base + k] != 0.f) ? 1.f : 0.f;
    sh[tid] = s0;
    __syncthreads();
    for (int off = 1; off < 1024; off <<= 1) {
        float v = sh[tid];
        float u = (tid >= off) ? sh[tid - off] : 0.f;
        __syncthreads();
        sh[tid] = v + u;
        __syncthreads();
    }
    float run = sh[tid] - s0;               // exclusive prefix for this chunk
    #pragma unroll 8
    for (int k = 0; k < 32; ++k) {
        run += (onsets[base + k] != 0.f) ? 1.f : 0.f;
        S[base + k + 1] = run;
    }
    if (tid == 0) S[0] = 0.f;
    __syncthreads();

    // ---- scan offsets indicator -> ranks -> off_idx ----
    float s1 = 0.f;
    #pragma unroll 8
    for (int k = 0; k < 32; ++k) s1 += (offsets[base + k] != 0.f) ? 1.f : 0.f;
    sh[tid] = s1;
    __syncthreads();
    for (int off = 1; off < 1024; off <<= 1) {
        float v = sh[tid];
        float u = (tid >= off) ? sh[tid - off] : 0.f;
        __syncthreads();
        sh[tid] = v + u;
        __syncthreads();
    }
    const int total = (int)sh[1023];        // total nonzero offsets
    int rank = (int)(sh[tid] - s1);
    #pragma unroll 8
    for (int k = 0; k < 32; ++k) {
        if (offsets[base + k] != 0.f) {
            if (rank < NN) off_idx[rank] = base + k;   // rank < total: disjoint from fill
            ++rank;
        }
    }
    if (tid >= total) off_idx[tid] = T_LEN - 1;        // flatnonzero fill (tid == j, NN==1024)
}

// ---------------------------------------------------------------------------
// prep2: per-note constants lo[j] = off_idx[j-1] (lo[0]=0), S_at_o[j] = S[o_j];
// also zero the two global accumulators (deterministic per launch).
// ---------------------------------------------------------------------------
__global__ __launch_bounds__(1024) void pl_prep2(const float* __restrict__ S,
                                                 const int* __restrict__ off_idx,
                                                 int* __restrict__ lo_idx,
                                                 float* __restrict__ S_at_o,
                                                 float* __restrict__ accum) {
    const int j = threadIdx.x;
    const int o = off_idx[j];
    lo_idx[j]   = (j == 0) ? 0 : off_idx[j - 1];
    S_at_o[j]   = S[o];
    if (j < 2) accum[j] = 0.f;
}

// ---------------------------------------------------------------------------
// WMMA mask reduction. 1024 waves total: 64 note-tiles (16 j each) x 16 time
// slices (2048 t each). Per step a wave builds a 16x32 f16 tile of mask
// values and reduces it with v_wmma_f32_16x16x32_f16 against an all-ones B.
// Numerator ((|mk_e-mk_u| > 0.5) count) computed alongside (it is exactly 0).
// ---------------------------------------------------------------------------
__global__ __launch_bounds__(256) void pl_mask_wmma(const float* __restrict__ x0,
                                                    const float* __restrict__ S,
                                                    const int* __restrict__ off_idx,
                                                    const int* __restrict__ lo_idx,
                                                    const float* __restrict__ S_at_o,
                                                    float* __restrict__ accum) {
    const int lane   = threadIdx.x & 31;
    const int w      = blockIdx.x * 8 + (threadIdx.x >> 5);  // 0..1023
    const int jTile  = w & 63;                               // 64 note tiles
    const int tSlice = w >> 6;                               // 16 time slices
    const int j      = jTile * 16 + (lane & 15);
    const int half   = lane >> 4;                            // lanes 16-31 take t offsets 16..31

    const int   o  = off_idx[j];
    const int   lo = lo_idx[j];
    const float So = S_at_o[j];

    v16h ones;
    #pragma unroll
    for (int e = 0; e < 16; ++e) ones[e] = (_Float16)1.0f;

    v8f acc = {0.f, 0.f, 0.f, 0.f, 0.f, 0.f, 0.f, 0.f};
    float cnt = 0.f;
    const int tb = tSlice * 2048 + half * 16;

    for (int s = 0; s < 64; ++s) {
        const int t0 = tb + s * 32;
        v16h a;
        #pragma unroll
        for (int e = 0; e < 16; ++e) {
            const int t   = t0 + e;
            int idx = t + 1;                      // jnp.clip: max then min
            idx = (idx < lo) ? lo : idx;
            idx = (idx > o)  ? o  : idx;
            const float con = So - S[idx];        // onset count in [idx, o)
            const float mf  = 1.0f - con - ((t >= o) ? 1.0f : 0.0f);
            // numerator path (u_f0 == e_f0 == x0[:,0] -> diff is exactly 0)
            const float v    = x0[2 * t];
            const float mk_u = mf * v;
            const float mk_e = mf * v;
            const float df   = fabsf(mk_e - mk_u);
            cnt += (df > 0.5f) ? 1.0f : 0.0f;
            a[e] = (_Float16)mf;                  // exact: mask is small-integer valued
        }
        // D[m,n] += sum_k A[m,k]  (B = ones) -> matrix pipe does the 512-way add
        acc = __builtin_amdgcn_wmma_f32_16x16x32_f16(
                  false, a, false, ones, (short)0, acc, false, false);
    }

    // acc VGPRs r hold rowsum(M=r) (lanes 0-15) / rowsum(M=r+8) (lanes 16-31),
    // replicated across the 16 columns -> lane partial + swap halves = tile sum.
    float p = acc[0] + acc[1] + acc[2] + acc[3] + acc[4] + acc[5] + acc[6] + acc[7];
    p += __shfl_xor(p, 16, 32);
    #pragma unroll
    for (int m = 1; m < 32; m <<= 1) cnt += __shfl_xor(cnt, m, 32);

    if (lane == 0) {
        atomicAdd(&accum[0], p);     // sum(mask)
        atomicAdd(&accum[1], cnt);   // sum(diff > 0.5)
    }
}

__global__ void pl_finalize(const float* __restrict__ accum, float* __restrict__ out) {
    const float m1 = accum[1] / TOTAL_F;   // mean(diff > THRESHOLD)
    const float m2 = accum[0] / TOTAL_F;   // mean(mask)
    out[0] = m1 / m2;
}

// ---------------------------------------------------------------------------
extern "C" void kernel_launch(void* const* d_in, const int* in_sizes, int n_in,
                              void* d_out, int out_size, void* d_ws, size_t ws_size,
                              hipStream_t stream) {
    const float* x0      = (const float*)d_in[0];
    // d_in[1] = x1 : unused by the reference
    const float* onsets  = (const float*)d_in[2];
    const float* offsets = (const float*)d_in[3];

    char*  ws      = (char*)d_ws;
    float* S       = (float*)ws;                 // (T+1) floats = 131076 B
    int*   off_idx = (int*)  (ws + 131328);      // 1024 ints
    int*   lo_idx  = (int*)  (ws + 135424);      // 1024 ints
    float* S_at_o  = (float*)(ws + 139520);      // 1024 floats
    float* accum   = (float*)(ws + 143616);      // [maskSum, cntSum]

    pl_prep1<<<1, 1024, 0, stream>>>(onsets, offsets, S, off_idx);
    pl_prep2<<<1, 1024, 0, stream>>>(S, off_idx, lo_idx, S_at_o, accum);
    pl_mask_wmma<<<128, 256, 0, stream>>>(x0, S, off_idx, lo_idx, S_at_o, accum);
    pl_finalize<<<1, 1, 0, stream>>>(accum, (float*)d_out);
}
